// OWAConv_fm_69604239999283
// MI455X (gfx1250) — matrix operand order repaired and checked by reading, not compiled, not run on hardware
//
#include <hip/hip_runtime.h>
#include <hip/hip_bf16.h>

// ---------------------------------------------------------------------------
// OWA conv  (B,H,W,C)=(32,56,56,256), F=256
//   sums[b,c] = sum_{hw} in[b,hw,c]
//   ranks     = argsort(argsort(sums)); idx = ranks[:, ::-1]
//   out[b,hw,f] = sum_c in[b,hw,idx[b,c]] * K[c,f]
// Reformulated:  out = sum_j in[:,j] * K[pos_b[j], :]   (pos = inverse of idx)
// Per-batch permuted kernel is precomputed in WMMA B-fragment layout (bf16
// hi/lo planes) -> GEMM has no LDS. A is split hi/lo in registers with cheap
// truncation + v_perm_b32 / v_fma_mix packing; each 16x16x32 step issues 3
// v_wmma_f32_16x16x32_bf16 (hi*hi + lo*hi + hi*lo) for ~f32 accuracy.
// Wave tile 32(M) x 256(F): A loaded/converted once per element, each B
// fragment pair feeds 6 WMMAs; 96 WMMAs per K-chunk -> matrix-pipe bound.
// ---------------------------------------------------------------------------

#define HW   3136   // 56*56
#define C_   256
#define F_   256

typedef __attribute__((ext_vector_type(16))) __bf16          v16bf;
typedef __attribute__((ext_vector_type(8)))  unsigned int    v8u;
typedef __attribute__((ext_vector_type(8)))  float           v8f;

__device__ __forceinline__ unsigned short f32_to_bf16_rne(float f) {
  unsigned int u = __builtin_bit_cast(unsigned int, f);
  unsigned int r = u + 0x7FFFu + ((u >> 16) & 1u);
  return (unsigned short)(r >> 16);
}
__device__ __forceinline__ float bf16_to_f32(unsigned short h) {
  unsigned int u = ((unsigned int)h) << 16;
  return __builtin_bit_cast(float, u);
}

// ---------------------------------------------------------------------------
// Kernel 1: per-(batch,channel) sums over H*W.  grid = (16 chunks, 32 batches)
// ---------------------------------------------------------------------------
__global__ __launch_bounds__(256) void owa_sums(const float* __restrict__ inp,
                                                float* __restrict__ sums) {
  const int b     = blockIdx.y;
  const int chunk = blockIdx.x;           // 16 chunks of 196 rows
  const int c     = threadIdx.x;          // channel
  const float* A  = inp + ((size_t)b * HW + (size_t)chunk * 196) * C_;
  float acc = 0.f;
  for (int r = 0; r < 196; ++r) acc += A[(size_t)r * C_ + c];
  atomicAdd(&sums[b * C_ + c], acc);
}

// ---------------------------------------------------------------------------
// Kernel 2: per-batch stable ranks -> inverse permutation -> permuted kernel
// written in WMMA B-fragment layout, bf16 hi/lo planes.
//   Kfrag[b][ft(16)][kc(8)][lane(32)][e(16)]:
//     lane = (f&15) + (kk&16), e = kk&15, kk = K index within 32-chunk.
//   => each lane's 16 bf16 (K ascending) are 32 contiguous bytes.
// grid = 32 blocks (one per batch).
// ---------------------------------------------------------------------------
__global__ __launch_bounds__(256) void owa_rank_perm(const float* __restrict__ sums,
                                                     const float* __restrict__ kern,
                                                     unsigned short* __restrict__ Kph,
                                                     unsigned short* __restrict__ Kpl) {
  const int b = blockIdx.x;
  const int t = threadIdx.x;              // output feature f
  __shared__ float s[C_];
  __shared__ int   pos[C_];
  s[t] = sums[b * C_ + t];
  __syncthreads();
  const float sv = s[t];
  int rank = 0;
  for (int c2 = 0; c2 < C_; ++c2) {
    const float s2 = s[c2];
    rank += (s2 < sv || (s2 == sv && c2 < t)) ? 1 : 0;   // stable argsort rank
  }
  // idx[c] = rank[C-1-c]  =>  pos[rank[c']] = C-1-c'
  pos[rank] = (C_ - 1) - t;
  __syncthreads();

  const int ft = t >> 4;                  // F tile
  const int fl = t & 15;                  // lane low bits
  for (int j = 0; j < C_; ++j) {          // j = GEMM K index
    const int   src = pos[j];
    const float val = kern[src * F_ + t]; // coalesced along f
    const unsigned short h = f32_to_bf16_rne(val);
    const unsigned short l = f32_to_bf16_rne(val - bf16_to_f32(h));
    const int kci  = j >> 5;
    const int jj   = j & 31;
    const int lane = fl + (jj & 16);      // +16 for K-half 16..31
    const int e    = jj & 15;
    const size_t o = ((((size_t)b * 16 + ft) * 8 + kci) * 32 + lane) * 16 + e;
    Kph[o] = h;
    Kpl[o] = l;
  }
}

// ---------------------------------------------------------------------------
// A fragment builder: 16 f32 -> bf16 hi/lo truncation split in fragment order.
// ---------------------------------------------------------------------------
__device__ __forceinline__ void split_a(const float* __restrict__ Arow, int kc,
                                        int khalf, v16bf& ah, v16bf& al) {
  v8u ahu, alu;
#pragma unroll
  for (int v = 0; v < 8; ++v) {
    const int k0 = kc + khalf + v * 2;
    const float2 f2 = *(const float2*)&Arow[k0];
    const unsigned int u0 = __builtin_bit_cast(unsigned int, f2.x);
    const unsigned int u1 = __builtin_bit_cast(unsigned int, f2.y);
    ahu[v] = __builtin_amdgcn_perm(u1, u0, 0x07060302u);   // [u0.hi16 | u1.hi16]
    const float l0 = f2.x - __builtin_bit_cast(float, u0 & 0xffff0000u);
    const float l1 = f2.y - __builtin_bit_cast(float, u1 & 0xffff0000u);
    alu[v] = __builtin_amdgcn_perm(__builtin_bit_cast(unsigned int, l1),
                                   __builtin_bit_cast(unsigned int, l0),
                                   0x07060302u);
  }
  ah = __builtin_bit_cast(v16bf, ahu);
  al = __builtin_bit_cast(v16bf, alu);
}

// ---------------------------------------------------------------------------
// Kernel 3: GEMM  out[b] = A[b](3136x256) * Kperm[b](256x256), no LDS.
// block = 256 thr (8 waves); block tile 256(M) x 256(F); wave tile 32 x 256.
// grid = (13, 32);  3136 = 98*32, tail block has 2 active waves.
// ---------------------------------------------------------------------------
__global__ __launch_bounds__(256) void owa_gemm(const float* __restrict__ inp,
                                                const unsigned short* __restrict__ Kph,
                                                const unsigned short* __restrict__ Kpl,
                                                float* __restrict__ out) {
  const int b    = blockIdx.y;
  const int tid  = threadIdx.x;
  const int lane = tid & 31;
  const int m0   = blockIdx.x * 256 + (tid >> 5) * 32;
  if (m0 >= HW) return;                   // wave-uniform, no LDS/barriers

  // two A fragments: rows [m0, m0+16) and [m0+16, m0+32)
  const float* A0 = inp + ((size_t)b * HW + m0 + (lane & 15)) * C_;
  const float* A1 = A0 + (size_t)16 * C_;
  // fragment base: [b][ft][kci][lane][e]; this wave uses all 16 F tiles
  const size_t fragBase = ((size_t)b * 16 * 8) * (32 * 16) + (size_t)lane * 16;
  const unsigned short* Bh = Kph + fragBase;
  const unsigned short* Bl = Kpl + fragBase;

  const int khalf = (lane >= 16) ? 16 : 0;  // A-layout half-wave K shift

  v8f acc0[16];
  v8f acc1[16];
#pragma unroll
  for (int n = 0; n < 16; ++n) { acc0[n] = v8f{}; acc1[n] = v8f{}; }

#pragma unroll
  for (int kci = 0; kci < 8; ++kci) {
    const int kc = kci * 32;
    v16bf ah0, al0, ah1, al1;
    split_a(A0, kc, khalf, ah0, al0);
    split_a(A1, kc, khalf, ah1, al1);

#pragma unroll
    for (int n = 0; n < 16; ++n) {
      // fragment (ft = n, kci): contiguous 32B per lane
      const size_t fo = ((size_t)n * 8 + kci) * (32 * 16);
      const v16bf bh = *(const v16bf*)&Bh[fo];
      const v16bf bl = *(const v16bf*)&Bl[fo];
      // (ah+al)*(bh+bl) ~= ah*bh + al*bh + ah*bl   (f32 accumulate)
      acc0[n] = __builtin_amdgcn_wmma_f32_16x16x32_bf16(false, ah0, false, bh,
                                                        (short)0, acc0[n], false, false);
      acc0[n] = __builtin_amdgcn_wmma_f32_16x16x32_bf16(false, al0, false, bh,
                                                        (short)0, acc0[n], false, false);
      acc0[n] = __builtin_amdgcn_wmma_f32_16x16x32_bf16(false, ah0, false, bl,
                                                        (short)0, acc0[n], false, false);
      acc1[n] = __builtin_amdgcn_wmma_f32_16x16x32_bf16(false, ah1, false, bh,
                                                        (short)0, acc1[n], false, false);
      acc1[n] = __builtin_amdgcn_wmma_f32_16x16x32_bf16(false, al1, false, bh,
                                                        (short)0, acc1[n], false, false);
      acc1[n] = __builtin_amdgcn_wmma_f32_16x16x32_bf16(false, ah1, false, bl,
                                                        (short)0, acc1[n], false, false);
    }
  }

  float* O = out + (size_t)b * HW * F_;
  const int orow = m0 + ((lane >= 16) ? 8 : 0);   // C/D layout: M = v + 8*(lane>=16)
  const int oc0  = lane & 15;
#pragma unroll
  for (int n = 0; n < 16; ++n) {
#pragma unroll
    for (int v = 0; v < 8; ++v) {
      O[(size_t)(orow + v) * F_ + oc0 + n * 16]      = acc0[n][v];
      O[(size_t)(orow + 16 + v) * F_ + oc0 + n * 16] = acc1[n][v];
    }
  }
}

// ---------------------------------------------------------------------------
extern "C" void kernel_launch(void* const* d_in, const int* in_sizes, int n_in,
                              void* d_out, int out_size, void* d_ws, size_t ws_size,
                              hipStream_t stream) {
  const float* inp  = (const float*)d_in[0];   // (32,56,56,256) f32
  const float* kern = (const float*)d_in[1];   // (1,1,256,256)  f32
  float*       out  = (float*)d_out;           // (32,56,56,256) f32

  // scratch layout
  float*          sums = (float*)d_ws;                                   // 32KB
  unsigned short* Kph  = (unsigned short*)((char*)d_ws + 32768);         // 4MB
  unsigned short* Kpl  = Kph + (size_t)32 * C_ * F_;                     // 4MB

  hipMemsetAsync(sums, 0, 32 * C_ * sizeof(float), stream);

  owa_sums<<<dim3(16, 32), 256, 0, stream>>>(inp, sums);
  owa_rank_perm<<<dim3(32), 256, 0, stream>>>(sums, kern, Kph, Kpl);
  owa_gemm<<<dim3(13, 32), 256, 0, stream>>>(inp, Kph, Kpl, out);
}
